// MultiHeadAttention_61168924230475
// MI455X (gfx1250) — matrix operand order repaired and checked by reading, not compiled, not run on hardware
//
#include <hip/hip_runtime.h>
#include <hip/hip_bf16.h>

// ---------------------------------------------------------------------------
// Types for CDNA5 WMMA (wave32)
// ---------------------------------------------------------------------------
typedef __attribute__((ext_vector_type(16))) __bf16 v16bf;
typedef __attribute__((ext_vector_type(8)))  __bf16 v8bf;
typedef __attribute__((ext_vector_type(8)))  float  v8f;

union BF16Frag {
    v16bf v;
    v8bf  h[2];
};

__device__ __forceinline__ v8f v8f_zero() {
    v8f z;
#pragma unroll
    for (int i = 0; i < 8; ++i) z[i] = 0.0f;
    return z;
}

__device__ __forceinline__ __bf16 f2bf(float f) {
    union { float f; unsigned u; } c; c.f = f;
    unsigned r = c.u + 0x7FFFu + ((c.u >> 16) & 1u);   // round-to-nearest-even
    union { unsigned short s; __bf16 b; } o;
    o.s = (unsigned short)(r >> 16);
    return o.b;
}

__device__ __forceinline__ v8f wmma_bf16(const v16bf& a, const v16bf& b, const v8f& c) {
    return __builtin_amdgcn_wmma_f32_16x16x32_bf16(false, a, false, b, (short)0, c, false, false);
}

// Problem constants
#define BSZ   4
#define TSEQ  2048
#define CEMB  1024
#define NHEAD 16
#define DHEAD 64
#define MTOK  (BSZ * TSEQ)   // 8192

// ---------------------------------------------------------------------------
// Kernel 1: x fp32 -> bf16 (row-major [M][C], A-side layout)
// ---------------------------------------------------------------------------
__global__ __launch_bounds__(256) void cvt_x_kernel(const float* __restrict__ x,
                                                    __bf16* __restrict__ xh) {
    int g = blockIdx.x * blockDim.x + threadIdx.x;           // one float4 each
    const float4* xv = (const float4*)x;
    float4 v = xv[g];
    union { __bf16 b[4]; uint2 u; } pk;
    pk.b[0] = f2bf(v.x); pk.b[1] = f2bf(v.y);
    pk.b[2] = f2bf(v.z); pk.b[3] = f2bf(v.w);
    *(uint2*)(xh + (size_t)g * 4) = pk.u;
}

// ---------------------------------------------------------------------------
// Kernel 2: weight fp32 [k][n] -> bf16 transposed Wt[n][k] (B-side layout)
// ---------------------------------------------------------------------------
__global__ __launch_bounds__(256) void cvt_wt_kernel(const float* __restrict__ W,
                                                     __bf16* __restrict__ Wt) {
    __shared__ __bf16 tile[32][33];
    int tx = threadIdx.x;            // 0..31
    int ty = threadIdx.y;            // 0..7
    int n0 = blockIdx.x * 32;
    int k0 = blockIdx.y * 32;
#pragma unroll
    for (int i = 0; i < 32; i += 8)
        tile[ty + i][tx] = f2bf(W[(size_t)(k0 + ty + i) * CEMB + n0 + tx]);
    __syncthreads();
#pragma unroll
    for (int i = 0; i < 32; i += 8)
        Wt[(size_t)(n0 + ty + i) * CEMB + k0 + tx] = tile[tx][ty + i];
}

// ---------------------------------------------------------------------------
// Shared tiled GEMM core: 128(M) x 256(N) block tile, 8 waves as 2x4,
// each wave computes a 64x64 sub-tile (4x4 WMMA frags), K-step 32.
// Structural double buffering: K-loop unrolled by 2 so both LDS buffers are
// compile-time constants (no cselect/branch bookkeeping); 1 barrier per step.
// A: [M][1024] row-major bf16.  Bt: [N][1024] row-major bf16 (pre-transposed).
// ---------------------------------------------------------------------------
#define LDT 40   // LDS row stride in bf16 (32 + 8 pad, keeps 16B alignment)

struct Stage {
    uint4 a0, a1, b0, b1, b2, b3;
};

__device__ __forceinline__ Stage stage_load(const __bf16* sa, const __bf16* sb) {
    Stage s;
    s.a0 = *(const uint4*)(sa);
    s.a1 = *(const uint4*)(sa + 8);
    s.b0 = *(const uint4*)(sb);
    s.b1 = *(const uint4*)(sb + 8);
    s.b2 = *(const uint4*)(sb + 16);
    s.b3 = *(const uint4*)(sb + 24);
    return s;
}

__device__ __forceinline__ void stage_store(__bf16* da, __bf16* db, const Stage& s) {
    *(uint4*)(da)      = s.a0;
    *(uint4*)(da + 8)  = s.a1;
    *(uint4*)(db)      = s.b0;
    *(uint4*)(db + 8)  = s.b1;
    *(uint4*)(db + 16) = s.b2;
    *(uint4*)(db + 24) = s.b3;
}

__device__ __forceinline__ void compute_step(const __bf16* Acur, const __bf16* Bcur,
                                             v8f acc[4][4],
                                             int m_w, int n_w, int lr, int hi) {
    BF16Frag a[4], b[4];
#pragma unroll
    for (int i = 0; i < 4; ++i) {     // A frag: row M=lr, K split by half-wave
        const __bf16* ar = Acur + (m_w * 64 + i * 16 + lr) * LDT;
        a[i].h[0] = *(const v8bf*)(ar + 8 * hi);
        a[i].h[1] = *(const v8bf*)(ar + 16 + 8 * hi);
    }
#pragma unroll
    for (int j = 0; j < 4; ++j) {     // B frag: column n=lr, K split by half-wave
        const __bf16* br = Bcur + (n_w * 64 + j * 16 + lr) * LDT;
        b[j].h[0] = *(const v8bf*)(br + 8 * hi);
        b[j].h[1] = *(const v8bf*)(br + 16 + 8 * hi);
    }
#pragma unroll
    for (int i = 0; i < 4; ++i)
#pragma unroll
        for (int j = 0; j < 4; ++j)
            acc[i][j] = wmma_bf16(a[i].v, b[j].v, acc[i][j]);
}

__device__ __forceinline__ void gemm_core_128x256(
    const __bf16* __restrict__ A, const __bf16* __restrict__ Bt,
    int m0, int n0,
    __bf16* Ash0, __bf16* Ash1, __bf16* Bsh0, __bf16* Bsh1,
    v8f acc[4][4], int tid)
{
    const int lane = tid & 31;
    const int wave = tid >> 5;
    const int lr   = lane & 15;
    const int hi   = lane >> 4;
    const int m_w  = wave >> 2;      // 0..1
    const int n_w  = wave & 3;       // 0..3
    const int ra   = tid >> 1;       // 0..127 : A row loaded by this thread
    const int ka   = (tid & 1) * 16; // 0 / 16 : A K-offset loaded by this thread
    const int rb   = tid;            // 0..255 : B row loaded by this thread

    const __bf16* gA = A  + (size_t)(m0 + ra) * CEMB + ka;
    const __bf16* gB = Bt + (size_t)(n0 + rb) * CEMB;
    __bf16* dA0 = Ash0 + ra * LDT + ka;
    __bf16* dB0 = Bsh0 + rb * LDT;
    __bf16* dA1 = Ash1 + ra * LDT + ka;
    __bf16* dB1 = Bsh1 + rb * LDT;

    // preload K-tile 0 into buffer 0
    stage_store(dA0, dB0, stage_load(gA, gB));
    __syncthreads();

#pragma unroll 1
    for (int kk = 0; kk < CEMB - 64; kk += 64) {
        // ---- half 1: compute tile kk (buf0), stage tile kk+32 -> buf1 ----
        Stage s1 = stage_load(gA + kk + 32, gB + kk + 32);
        __builtin_prefetch(gA + kk + 96, 0, 3);   // speculative GL2 prefetch
        __builtin_prefetch(gB + kk + 96, 0, 3);
        compute_step(Ash0, Bsh0, acc, m_w, n_w, lr, hi);
        stage_store(dA1, dB1, s1);
        __syncthreads();
        // ---- half 2: compute tile kk+32 (buf1), stage tile kk+64 -> buf0 --
        Stage s2 = stage_load(gA + kk + 64, gB + kk + 64);
        __builtin_prefetch(gA + kk + 128, 0, 3);
        __builtin_prefetch(gB + kk + 128, 0, 3);
        compute_step(Ash1, Bsh1, acc, m_w, n_w, lr, hi);
        stage_store(dA0, dB0, s2);
        __syncthreads();
    }
    // tail: tiles CEMB-64 (buf0) and CEMB-32 (staged -> buf1)
    {
        Stage s1 = stage_load(gA + CEMB - 32, gB + CEMB - 32);
        compute_step(Ash0, Bsh0, acc, m_w, n_w, lr, hi);
        stage_store(dA1, dB1, s1);
        __syncthreads();
        compute_step(Ash1, Bsh1, acc, m_w, n_w, lr, hi);
    }
}

// ---------------------------------------------------------------------------
// Kernel 3: fused QKV projection.
//   grid = (64 Mtiles, 12) ; blockIdx.y: which = y/4 (0=Q,1=K,2=V), ntile=y%4
//   Q,K written as bf16 [B,H,T,d] ; V written transposed bf16 [B,H,d,T]
// ---------------------------------------------------------------------------
__global__ __launch_bounds__(256) void qkv_gemm_kernel(
    const __bf16* __restrict__ xh,
    const __bf16* __restrict__ wqt, const __bf16* __restrict__ wkt,
    const __bf16* __restrict__ wvt,
    __bf16* __restrict__ q_ws, __bf16* __restrict__ k_ws, __bf16* __restrict__ v_ws)
{
    __shared__ __bf16 Ash[2][128 * LDT];
    __shared__ __bf16 Bsh[2][256 * LDT];

    const int tid   = threadIdx.x;
    const int which = blockIdx.y >> 2;            // 0=Q 1=K 2=V
    const int n0    = (blockIdx.y & 3) * 256;
    const int m0    = blockIdx.x * 128;
    const __bf16* Bt = (which == 0) ? wqt : (which == 1) ? wkt : wvt;

    v8f acc[4][4];
#pragma unroll
    for (int i = 0; i < 4; ++i)
#pragma unroll
        for (int j = 0; j < 4; ++j) acc[i][j] = v8f_zero();

    gemm_core_128x256(xh, Bt, m0, n0, Ash[0], Ash[1], Bsh[0], Bsh[1], acc, tid);

    const int lane = tid & 31, wave = tid >> 5;
    const int lr = lane & 15, hi = lane >> 4;
    const int m_w = wave >> 2, n_w = wave & 3;

    if (which == 2) {   // V, stored transposed [B,H,d,T]
#pragma unroll
        for (int i = 0; i < 4; ++i)
#pragma unroll
            for (int j = 0; j < 4; ++j)
#pragma unroll
                for (int r = 0; r < 8; ++r) {
                    int gm = m0 + m_w * 64 + i * 16 + r + 8 * hi;
                    int gn = n0 + n_w * 64 + j * 16 + lr;
                    int b = gm >> 11, t = gm & (TSEQ - 1);
                    int h = gn >> 6,  dd = gn & (DHEAD - 1);
                    v_ws[(((size_t)(b * NHEAD + h) * DHEAD + dd) * TSEQ) + t] =
                        f2bf(acc[i][j][r]);
                }
    } else {            // Q or K, stored [B,H,T,d]
        __bf16* dst = which ? k_ws : q_ws;
#pragma unroll
        for (int i = 0; i < 4; ++i)
#pragma unroll
            for (int j = 0; j < 4; ++j)
#pragma unroll
                for (int r = 0; r < 8; ++r) {
                    int gm = m0 + m_w * 64 + i * 16 + r + 8 * hi;
                    int gn = n0 + n_w * 64 + j * 16 + lr;
                    int b = gm >> 11, t = gm & (TSEQ - 1);
                    int h = gn >> 6,  dd = gn & (DHEAD - 1);
                    dst[(((size_t)(b * NHEAD + h) * TSEQ + t) * DHEAD) + dd] =
                        f2bf(acc[i][j][r]);
                }
    }
}

// ---------------------------------------------------------------------------
// Kernel 4: causal flash attention.
//   grid = (T/64 query tiles, B*H) ; block = 128 (4 waves, 16 q-rows each)
//   S = Q Kᵀ (bf16 WMMA, f32 acc), online softmax, O += P V.
// ---------------------------------------------------------------------------
#define LDK 72   // LDS row stride in bf16 for 64-wide tiles (64 + 8 pad)

__global__ __launch_bounds__(128) void attn_kernel(
    const __bf16* __restrict__ q_ws, const __bf16* __restrict__ k_ws,
    const __bf16* __restrict__ v_ws, __bf16* __restrict__ y_ws)
{
    __shared__ __bf16 Kt[64 * LDK];          // [key][d]
    __shared__ __bf16 Vt[64 * LDK];          // [d][key]   (from V stored [d][T])
    __shared__ __bf16 Pt[4][16 * LDK];       // per-wave P scratch [qrow][key]

    const int bh   = blockIdx.y;             // 0..63  (= b*16 + h)
    const int qt   = blockIdx.x;             // 0..31
    const int tid  = threadIdx.x;
    const int wave = tid >> 5, lane = tid & 31;
    const int lr = lane & 15, hi = lane >> 4;
    const size_t qk_base = (size_t)bh * TSEQ * DHEAD;
    const size_t v_base  = (size_t)bh * DHEAD * TSEQ;
    const int q0 = qt * 64;

    // Q A-fragments (2 K-chunks of 32 over d=64), loaded once from global
    BF16Frag aq[2];
    {
        const __bf16* qrow = q_ws + qk_base + (size_t)(q0 + wave * 16 + lr) * DHEAD;
#pragma unroll
        for (int c = 0; c < 2; ++c) {
            aq[c].h[0] = *(const v8bf*)(qrow + 32 * c + 8 * hi);
            aq[c].h[1] = *(const v8bf*)(qrow + 32 * c + 16 + 8 * hi);
        }
    }

    v8f o[4];
#pragma unroll
    for (int j = 0; j < 4; ++j) o[j] = v8f_zero();
    float m_i[8], l_i[8];
#pragma unroll
    for (int r = 0; r < 8; ++r) { m_i[r] = -1e30f; l_i[r] = 0.0f; }
    const float scale = 0.125f;              // 1/sqrt(64)

    const int ldr = tid >> 1;                // 0..63
    const int lko = (tid & 1) * 32;          // 0 or 32 (elements)

#pragma unroll 1
    for (int kt = 0; kt <= qt; ++kt) {
        const int k0 = kt * 64;
        __syncthreads();                     // LDS reuse across iterations
        {   // stage K tile [64 keys][64 d] and V tile [64 d][64 keys]
            const __bf16* src = k_ws + qk_base + (size_t)(k0 + ldr) * DHEAD + lko;
            __bf16* dst = Kt + ldr * LDK + lko;
            *(uint4*)(dst)      = *(const uint4*)(src);
            *(uint4*)(dst + 8)  = *(const uint4*)(src + 8);
            *(uint4*)(dst + 16) = *(const uint4*)(src + 16);
            *(uint4*)(dst + 24) = *(const uint4*)(src + 24);
            src = v_ws + v_base + (size_t)ldr * TSEQ + k0 + lko;
            dst = Vt + ldr * LDK + lko;
            *(uint4*)(dst)      = *(const uint4*)(src);
            *(uint4*)(dst + 8)  = *(const uint4*)(src + 8);
            *(uint4*)(dst + 16) = *(const uint4*)(src + 16);
            *(uint4*)(dst + 24) = *(const uint4*)(src + 24);
        }
        __syncthreads();

        // S = Q Kᵀ  : 16(q) x 64(keys) per wave
        v8f s[4];
#pragma unroll
        for (int j = 0; j < 4; ++j) s[j] = v8f_zero();
#pragma unroll
        for (int c = 0; c < 2; ++c)
#pragma unroll
            for (int j = 0; j < 4; ++j) {
                BF16Frag bk;
                const __bf16* kr = Kt + (j * 16 + lr) * LDK;   // column = key
                bk.h[0] = *(const v8bf*)(kr + 32 * c + 8 * hi);
                bk.h[1] = *(const v8bf*)(kr + 32 * c + 16 + 8 * hi);
                s[j] = wmma_bf16(aq[c].v, bk.v, s[j]);
            }

        // scale + causal mask (gk > gq -> -inf); rows r live at M = r + 8*hi
        const int gq_base = q0 + wave * 16 + 8 * hi;
#pragma unroll
        for (int j = 0; j < 4; ++j) {
            const int gk = k0 + j * 16 + lr;
#pragma unroll
            for (int r = 0; r < 8; ++r)
                s[j][r] = (gk > gq_base + r) ? -1e30f : s[j][r] * scale;
        }

        // online softmax: row max reduced over the 16 lanes of each half-wave
        float mnew[8];
#pragma unroll
        for (int r = 0; r < 8; ++r) {
            mnew[r] = m_i[r];
#pragma unroll
            for (int j = 0; j < 4; ++j) mnew[r] = fmaxf(mnew[r], s[j][r]);
        }
#pragma unroll
        for (int off = 1; off < 16; off <<= 1)
#pragma unroll
            for (int r = 0; r < 8; ++r)
                mnew[r] = fmaxf(mnew[r], __shfl_xor(mnew[r], off, 32));

        float alpha[8];
#pragma unroll
        for (int r = 0; r < 8; ++r) {
            alpha[r] = __expf(m_i[r] - mnew[r]);
            m_i[r] = mnew[r];
        }
#pragma unroll
        for (int j = 0; j < 4; ++j)
#pragma unroll
            for (int r = 0; r < 8; ++r)
                s[j][r] = __expf(s[j][r] - mnew[r]);
#pragma unroll
        for (int r = 0; r < 8; ++r) {
            float part = s[0][r] + s[1][r] + s[2][r] + s[3][r];
            l_i[r] = l_i[r] * alpha[r] + part;     // lane-partial; reduced at end
        }
#pragma unroll
        for (int j = 0; j < 4; ++j)
#pragma unroll
            for (int r = 0; r < 8; ++r) o[j][r] *= alpha[r];

        // P: C-layout -> LDS -> A-layout (same-wave RAW: explicit DS wait)
#pragma unroll
        for (int j = 0; j < 4; ++j)
#pragma unroll
            for (int r = 0; r < 8; ++r)
                Pt[wave][(r + 8 * hi) * LDK + j * 16 + lr] = f2bf(s[j][r]);
        asm volatile("s_wait_dscnt 0" ::: "memory");

#pragma unroll
        for (int c = 0; c < 2; ++c) {
            BF16Frag ap;
            const __bf16* pr = Pt[wave] + lr * LDK;
            ap.h[0] = *(const v8bf*)(pr + 32 * c + 8 * hi);
            ap.h[1] = *(const v8bf*)(pr + 32 * c + 16 + 8 * hi);
#pragma unroll
            for (int j = 0; j < 4; ++j) {
                BF16Frag bv;
                const __bf16* vr = Vt + (j * 16 + lr) * LDK;   // column = d index
                bv.h[0] = *(const v8bf*)(vr + 32 * c + 8 * hi);
                bv.h[1] = *(const v8bf*)(vr + 32 * c + 16 + 8 * hi);
                o[j] = wmma_bf16(ap.v, bv.v, o[j]);
            }
        }
    }

    // finalize: reduce l over 16 columns, normalize, store y [B,T,C] bf16
#pragma unroll
    for (int off = 1; off < 16; off <<= 1)
#pragma unroll
        for (int r = 0; r < 8; ++r) l_i[r] += __shfl_xor(l_i[r], off, 32);

    const int b = bh >> 4, h = bh & (NHEAD - 1);
#pragma unroll
    for (int j = 0; j < 4; ++j)
#pragma unroll
        for (int r = 0; r < 8; ++r) {
            int gq = q0 + wave * 16 + r + 8 * hi;
            float val = o[j][r] / l_i[r];
            y_ws[((size_t)(b * TSEQ + gq)) * CEMB + h * DHEAD + j * 16 + lr] = f2bf(val);
        }
}

// ---------------------------------------------------------------------------
// Kernel 5: output projection  out = y @ Wp + bp  (fp32 epilogue)
// ---------------------------------------------------------------------------
__global__ __launch_bounds__(256) void proj_gemm_kernel(
    const __bf16* __restrict__ yh, const __bf16* __restrict__ wpt,
    const float* __restrict__ bp, float* __restrict__ out)
{
    __shared__ __bf16 Ash[2][128 * LDT];
    __shared__ __bf16 Bsh[2][256 * LDT];

    const int tid = threadIdx.x;
    const int m0 = blockIdx.x * 128;
    const int n0 = blockIdx.y * 256;

    v8f acc[4][4];
#pragma unroll
    for (int i = 0; i < 4; ++i)
#pragma unroll
        for (int j = 0; j < 4; ++j) acc[i][j] = v8f_zero();

    gemm_core_128x256(yh, wpt, m0, n0, Ash[0], Ash[1], Bsh[0], Bsh[1], acc, tid);

    const int lane = tid & 31, wave = tid >> 5;
    const int lr = lane & 15, hi = lane >> 4;
    const int m_w = wave >> 2, n_w = wave & 3;
#pragma unroll
    for (int i = 0; i < 4; ++i)
#pragma unroll
        for (int j = 0; j < 4; ++j) {
            const int gn = n0 + n_w * 64 + j * 16 + lr;
            const float bias = bp[gn];
#pragma unroll
            for (int r = 0; r < 8; ++r) {
                const int gm = m0 + m_w * 64 + i * 16 + r + 8 * hi;
                out[(size_t)gm * CEMB + gn] = acc[i][j][r] + bias;
            }
        }
}

// ---------------------------------------------------------------------------
// Launcher
// ---------------------------------------------------------------------------
extern "C" void kernel_launch(void* const* d_in, const int* in_sizes, int n_in,
                              void* d_out, int out_size, void* d_ws, size_t ws_size,
                              hipStream_t stream) {
    const float* x  = (const float*)d_in[0];
    const float* Wq = (const float*)d_in[1];
    const float* Wk = (const float*)d_in[2];
    const float* Wv = (const float*)d_in[3];
    const float* Wp = (const float*)d_in[4];
    const float* bp = (const float*)d_in[5];
    float* out = (float*)d_out;

    char* ws = (char*)d_ws;
    const size_t MB = 1u << 20;
    __bf16* xh  = (__bf16*)(ws + 0 * MB);    // 16 MB  [8192][1024]
    __bf16* wqt = (__bf16*)(ws + 16 * MB);   //  2 MB  [n][k]
    __bf16* wkt = (__bf16*)(ws + 18 * MB);
    __bf16* wvt = (__bf16*)(ws + 20 * MB);
    __bf16* wpt = (__bf16*)(ws + 22 * MB);
    __bf16* qb  = (__bf16*)(ws + 24 * MB);   // 16 MB  [B,H,T,d]
    __bf16* kb  = (__bf16*)(ws + 40 * MB);   // 16 MB  [B,H,T,d]
    __bf16* vb  = (__bf16*)(ws + 56 * MB);   // 16 MB  [B,H,d,T]
    __bf16* yb  = (__bf16*)(ws + 72 * MB);   // 16 MB  [B*T][C]

    // 1) x -> bf16
    cvt_x_kernel<<<(MTOK * CEMB) / (256 * 4), 256, 0, stream>>>(x, xh);
    // 2) weights -> bf16, transposed to [n][k]
    dim3 wgrid(CEMB / 32, CEMB / 32), wblk(32, 8);
    cvt_wt_kernel<<<wgrid, wblk, 0, stream>>>(Wq, wqt);
    cvt_wt_kernel<<<wgrid, wblk, 0, stream>>>(Wk, wkt);
    cvt_wt_kernel<<<wgrid, wblk, 0, stream>>>(Wv, wvt);
    cvt_wt_kernel<<<wgrid, wblk, 0, stream>>>(Wp, wpt);
    // 3) fused QKV projection  (N = 3*1024 in tiles of 256)
    qkv_gemm_kernel<<<dim3(MTOK / 128, 12), 256, 0, stream>>>(xh, wqt, wkt, wvt,
                                                              qb, kb, vb);
    // 4) causal flash attention
    attn_kernel<<<dim3(TSEQ / 64, BSZ * NHEAD), 128, 0, stream>>>(qb, kb, vb, yb);
    // 5) output projection + bias
    proj_gemm_kernel<<<dim3(MTOK / 128, CEMB / 256), 256, 0, stream>>>(yb, wpt, bp, out);
}
